// DDM_22978075033940
// MI455X (gfx1250) — compile-verified, matrix operand order checked by reading, not run on hardware
//
#include <hip/hip_runtime.h>
#include <hip/hip_bf16.h>

// ---------------------------------------------------------------------------
// Problem constants (match reference)
// ---------------------------------------------------------------------------
#define NB_   8
#define NPER_ 4096
#define NN    (NB_ * NPER_)    // 32768 nodes
#define EE    262144           // edges
#define DD    256              // hidden
#define HH    4                // heads
#define DHH   64               // per-head dim
#define INID  128              // in/out dim
#define NBLK  6                // residual GAT blocks

typedef _Float16 half_t;
typedef __attribute__((ext_vector_type(16))) _Float16 v16h;
typedef __attribute__((ext_vector_type(8)))  _Float16 v8h;
typedef __attribute__((ext_vector_type(8)))  float    v8f;

__device__ __forceinline__ float gelu_f(float x) {
    return 0.5f * x * (1.0f + erff(x * 0.70710678118654752440f));
}

// monotone encoding of float into uint for atomicMax-based segment max
__device__ __forceinline__ unsigned encf(float f) {
    unsigned u = __float_as_uint(f);
    return (u & 0x80000000u) ? ~u : (u | 0x80000000u);
}
__device__ __forceinline__ float decf(unsigned u) {
    u = (u & 0x80000000u) ? (u & 0x7FFFFFFFu) : ~u;
    return __uint_as_float(u);
}

// ---------------------------------------------------------------------------
// WMMA GEMM: C[M,Nc] = act(A[M,K](f16) @ W[K,Nc](f16) + bias) (+res)
// block tile 128x64, 8 waves, each wave a 32x32 patch (2x2 wmma tiles)
// Requires: M%128==0, Nc%64==0, K%32==0 (true for all call sites)
// ---------------------------------------------------------------------------
template<int ACT, bool OUTF, bool OUTH, bool RES>
__global__ __launch_bounds__(256)
void k_gemm(const half_t* __restrict__ A, const half_t* __restrict__ W,
            const float* __restrict__ bias,
            float* __restrict__ outF, half_t* __restrict__ outH,
            const float* __restrict__ res,
            int M, int K, int Nc)
{
    constexpr int BM = 128, BN = 64, BK = 32, LDA = 40, LDB = 40;
    __shared__ half_t lA[BM * LDA];   // 10240 B
    __shared__ half_t lB[BN * LDB];   // 5120 B  (stored transposed: [n][k])

    const int m0   = blockIdx.x * BM;
    const int n0   = blockIdx.y * BN;
    const int tid  = threadIdx.x;
    const int lane = tid & 31;
    const int wave = tid >> 5;
    const int wm   = wave >> 1;       // 0..3  (32-row strip)
    const int wn   = wave & 1;        // 0..1  (32-col strip)
    const int hsel = lane >> 4;       // K-octet select per ISA A layout
    const int r16  = lane & 15;

    v8f acc[2][2] = {};

    const int ar = tid >> 1;                  // A stage row 0..127
    const int ac = (tid & 1) * 16;            // A stage col 0 or 16
    const int bk = tid >> 3;                  // B stage k 0..31
    const int bn = (tid & 7) * 8;             // B stage n 0..56

    for (int kk = 0; kk < K; kk += BK) {
        // ---- stage A tile (128 x 32 halves), contiguous 32B per thread ----
        {
            const half_t* src = A + (size_t)(m0 + ar) * K + kk + ac;
            uint4 p0 = *(const uint4*)(src);
            uint4 p1 = *(const uint4*)(src + 8);
            *(uint4*)(&lA[ar * LDA + ac])     = p0;
            *(uint4*)(&lA[ar * LDA + ac + 8]) = p1;
            if (kk + BK < K)  // gfx1250 global_prefetch_b8 on next A tile
                __builtin_prefetch(src + BK, 0, 1);
        }
        // ---- stage B tile (32k x 64n), written transposed [n][k] ----
        {
            const half_t* src = W + (size_t)(kk + bk) * Nc + n0 + bn;
            v8h row = *(const v8h*)(src);
            #pragma unroll
            for (int j = 0; j < 8; ++j) lB[(bn + j) * LDB + bk] = row[j];
        }
        __syncthreads();

        // ---- fragments (two b128 ds loads each, ISA 16-bit A/B layout) ----
        v16h af[2], bf[2];
        #pragma unroll
        for (int mi = 0; mi < 2; ++mi) {
            const half_t* p = &lA[(wm * 32 + mi * 16 + r16) * LDA + hsel * 8];
            v8h lo = *(const v8h*)(p);
            v8h hi = *(const v8h*)(p + 16);
            #pragma unroll
            for (int j = 0; j < 8; ++j) { af[mi][j] = lo[j]; af[mi][8 + j] = hi[j]; }
        }
        #pragma unroll
        for (int ni = 0; ni < 2; ++ni) {
            const half_t* p = &lB[(wn * 32 + ni * 16 + r16) * LDB + hsel * 8];
            v8h lo = *(const v8h*)(p);
            v8h hi = *(const v8h*)(p + 16);
            #pragma unroll
            for (int j = 0; j < 8; ++j) { bf[ni][j] = lo[j]; bf[ni][8 + j] = hi[j]; }
        }
        #pragma unroll
        for (int mi = 0; mi < 2; ++mi)
            #pragma unroll
            for (int ni = 0; ni < 2; ++ni)
                acc[mi][ni] = __builtin_amdgcn_wmma_f32_16x16x32_f16(
                    false, af[mi], false, bf[ni], (short)0, acc[mi][ni], false, false);
        __syncthreads();
    }

    // ---- epilogue: C layout VGPR r -> (M = r or 8+r), N = lane&15 ----
    #pragma unroll
    for (int mi = 0; mi < 2; ++mi)
        #pragma unroll
        for (int ni = 0; ni < 2; ++ni) {
            const int ncol = n0 + wn * 32 + ni * 16 + r16;
            const float bv = bias[ncol];
            #pragma unroll
            for (int r = 0; r < 8; ++r) {
                const int mrow = m0 + wm * 32 + mi * 16 + (hsel ? (8 + r) : r);
                const size_t idx = (size_t)mrow * Nc + ncol;
                float v = acc[mi][ni][r] + bv;
                if (ACT == 1) v = gelu_f(v);
                if (RES)  v += res[idx];
                if (OUTF) outF[idx] = v;
                if (OUTH) outH[idx] = (half_t)v;
            }
        }
}

// ---------------------------------------------------------------------------
// LayerNorm over D=256, one wave per row, writes f16 (GEMM A operand)
// ---------------------------------------------------------------------------
__global__ __launch_bounds__(256)
void k_layernorm_f16(const float* __restrict__ x, const float* __restrict__ g,
                     const float* __restrict__ b, half_t* __restrict__ y)
{
    const int row  = blockIdx.x * 8 + (threadIdx.x >> 5);
    const int lane = threadIdx.x & 31;
    const float* xr = x + (size_t)row * DD + lane * 8;
    float4 a0 = *(const float4*)(xr);
    float4 a1 = *(const float4*)(xr + 4);
    float v[8] = {a0.x, a0.y, a0.z, a0.w, a1.x, a1.y, a1.z, a1.w};

    float s = 0.f;
    #pragma unroll
    for (int j = 0; j < 8; ++j) s += v[j];
    #pragma unroll
    for (int off = 16; off > 0; off >>= 1) s += __shfl_xor(s, off, 32);
    const float mu = s * (1.0f / DD);

    float q = 0.f;
    #pragma unroll
    for (int j = 0; j < 8; ++j) { float d = v[j] - mu; q += d * d; }
    #pragma unroll
    for (int off = 16; off > 0; off >>= 1) q += __shfl_xor(q, off, 32);
    const float inv = rsqrtf(q * (1.0f / DD) + 1e-5f);

    half_t out[8];
    #pragma unroll
    for (int j = 0; j < 8; ++j) {
        const int c = lane * 8 + j;
        out[j] = (half_t)((v[j] - mu) * inv * g[c] + b[c]);
    }
    *(uint4*)(y + (size_t)row * DD + lane * 8) = *(const uint4*)out;
}

// ---------------------------------------------------------------------------
// Edge kernels (GATv2 attention), one wave per edge
// ---------------------------------------------------------------------------
__global__ __launch_bounds__(256)
void k_edge_logits(const float* __restrict__ fs, const float* __restrict__ fd,
                   const int* __restrict__ src, const int* __restrict__ dst,
                   const float* __restrict__ attn, float* __restrict__ logits,
                   unsigned* __restrict__ mEnc)
{
    const int e    = blockIdx.x * 8 + (threadIdx.x >> 5);
    const int lane = threadIdx.x & 31;
    const int s = src[e], d = dst[e];
    const float* ps = fs + (size_t)s * DD;
    const float* pd = fd + (size_t)d * DD;

    float acc[HH];
    #pragma unroll
    for (int h = 0; h < HH; ++h) {
        const int c = h * DHH + lane * 2;
        float2 a = *(const float2*)(ps + c);
        float2 q = *(const float2*)(pd + c);
        float2 w = *(const float2*)(attn + c);
        float e0 = a.x + q.x; e0 = (e0 > 0.f) ? e0 : 0.2f * e0;
        float e1 = a.y + q.y; e1 = (e1 > 0.f) ? e1 : 0.2f * e1;
        acc[h] = e0 * w.x + e1 * w.y;
    }
    #pragma unroll
    for (int off = 16; off > 0; off >>= 1)
        #pragma unroll
        for (int h = 0; h < HH; ++h) acc[h] += __shfl_xor(acc[h], off, 32);

    if (lane < HH) {
        float v = (lane == 0) ? acc[0] : (lane == 1) ? acc[1]
                : (lane == 2) ? acc[2] : acc[3];
        logits[(size_t)e * HH + lane] = v;
        atomicMax(&mEnc[(size_t)d * HH + lane], encf(v));
    }
}

__global__ __launch_bounds__(256)
void k_edge_exp(const float* __restrict__ logits, const unsigned* __restrict__ mEnc,
                const int* __restrict__ dst, float* __restrict__ aBuf,
                float* __restrict__ sBuf)
{
    const int i = blockIdx.x * 256 + threadIdx.x;    // E*H threads
    const int e = i >> 2, h = i & 3;
    const int d = dst[e];
    const float a = expf(logits[i] - decf(mEnc[(size_t)d * HH + h]));
    aBuf[i] = a;
    atomicAdd(&sBuf[(size_t)d * HH + h], a);
}

__global__ __launch_bounds__(256)
void k_edge_scatter(const float* __restrict__ fs, const int* __restrict__ src,
                    const int* __restrict__ dst, const float* __restrict__ aBuf,
                    const float* __restrict__ sBuf, float* __restrict__ agg)
{
    const int e    = blockIdx.x * 8 + (threadIdx.x >> 5);
    const int lane = threadIdx.x & 31;
    const int s = src[e], d = dst[e];
    #pragma unroll
    for (int h = 0; h < HH; ++h) {
        const float w = aBuf[(size_t)e * HH + h] / sBuf[(size_t)d * HH + h];
        const int c = h * DHH + lane * 2;
        float2 a = *(const float2*)(fs + (size_t)s * DD + c);
        atomicAdd(&agg[(size_t)d * DD + c],     w * a.x);
        atomicAdd(&agg[(size_t)d * DD + c + 1], w * a.y);
    }
}

// ---------------------------------------------------------------------------
// Elementwise helpers
// ---------------------------------------------------------------------------
__global__ void k_fill_u32(unsigned* p, unsigned v, int n) {
    int i = blockIdx.x * 256 + threadIdx.x;
    if (i < n) p[i] = v;
}

__global__ void k_gat_post(float* __restrict__ h, const float* __restrict__ agg,
                           const float* __restrict__ tc) {
    const int i = blockIdx.x * 256 + threadIdx.x;   // N*D
    const int n = i >> 8, c = i & 255, g = n >> 12; // NPER=4096
    h[i] = gelu_f(agg[i]) + tc[g * DD + c] + h[i];
}

__global__ void k_add(float* __restrict__ a, const float* __restrict__ b) {
    const int i = blockIdx.x * 256 + threadIdx.x;
    a[i] += b[i];
}

__global__ void k_concat_f16(const float* __restrict__ x_t,
                             const float* __restrict__ temb,
                             half_t* __restrict__ cat) {
    const int i = blockIdx.x * 256 + threadIdx.x;   // N*(IN+D)
    const int n = i / (INID + DD), c = i - n * (INID + DD);
    const float v = (c < INID) ? x_t[(size_t)n * INID + c]
                               : temb[(size_t)(n >> 12) * DD + (c - INID)];
    cat[i] = (half_t)v;
}

__global__ void k_f32_to_f16(const float* __restrict__ in, half_t* __restrict__ out, int n) {
    int i = blockIdx.x * 256 + threadIdx.x;
    if (i < n) out[i] = (half_t)in[i];
}

// ---------------------------------------------------------------------------
// Tiny time-embedding path (M=8): naive f32 kernels
// ---------------------------------------------------------------------------
__global__ void k_time_sin(const int* __restrict__ tidx, float* __restrict__ y) {
    const int i = blockIdx.x * 256 + threadIdx.x;   // 8*256
    const int row = i >> 8, c = i & 255;
    const float tf = (float)tidx[row] * 0.001f;
    const float k = -9.21034037197618f / 127.0f;    // -ln(10000)/(half-1)
    float v;
    if (c < 128) v = sinf(tf * expf((float)c * k));
    else         v = cosf(tf * expf((float)(c - 128) * k));
    y[i] = v;
}

template<int AIN, int AOUT>   // AIN: 0 none, 2 silu ; AOUT: 0 none, 1 gelu
__global__ void k_small_gemm(const float* __restrict__ X, const float* __restrict__ W,
                             const float* __restrict__ bias, float* __restrict__ Y,
                             int K, int Nc)
{
    const int i = blockIdx.x * 256 + threadIdx.x;   // 8*Nc threads
    const int row = i / Nc, col = i - row * Nc;
    float acc = bias[col];
    for (int k = 0; k < K; ++k) {
        float xv = X[row * K + k];
        if (AIN == 2) xv = xv / (1.0f + expf(-xv));
        acc += xv * W[(size_t)k * Nc + col];
    }
    if (AOUT == 1) acc = gelu_f(acc);
    Y[i] = acc;
}

// ---------------------------------------------------------------------------
// Host orchestration
// ---------------------------------------------------------------------------
extern "C" void kernel_launch(void* const* d_in, const int* in_sizes, int n_in,
                              void* d_out, int out_size, void* d_ws, size_t ws_size,
                              hipStream_t stream)
{
    // setup_inputs() dict order, params expanded in insertion order
    const float* x_t   = (const float*)d_in[0];
    const int*   tidx  = (const int*)  d_in[1];
    const int*   src   = (const int*)  d_in[2];
    const int*   dst   = (const int*)  d_in[3];
    const float* tp_w1 = (const float*)d_in[4];
    const float* tp_b1 = (const float*)d_in[5];
    const float* tp_w2 = (const float*)d_in[6];
    const float* tp_b2 = (const float*)d_in[7];
    const float* tm_w  = (const float*)d_in[8];
    const float* tm_b  = (const float*)d_in[9];
    const float* ip_w  = (const float*)d_in[10];
    const float* ip_b  = (const float*)d_in[11];
    const float* ln1_g = (const float*)d_in[12];
    const float* ln1_b = (const float*)d_in[13];
    const float* wsrc  = (const float*)d_in[14];
    const float* bsrc  = (const float*)d_in[15];
    const float* wdst  = (const float*)d_in[16];
    const float* bdst  = (const float*)d_in[17];
    const float* attn  = (const float*)d_in[18];
    const float* tw    = (const float*)d_in[19];
    const float* tb    = (const float*)d_in[20];
    const float* ln2_g = (const float*)d_in[21];
    const float* ln2_b = (const float*)d_in[22];
    const float* fw1   = (const float*)d_in[23];
    const float* fb1   = (const float*)d_in[24];
    const float* fw2   = (const float*)d_in[25];
    const float* fb2   = (const float*)d_in[26];
    const float* oln_g = (const float*)d_in[27];
    const float* oln_b = (const float*)d_in[28];
    const float* o_w1  = (const float*)d_in[29];
    const float* o_b1  = (const float*)d_in[30];
    const float* o_w2  = (const float*)d_in[31];
    const float* o_b2  = (const float*)d_in[32];
    (void)in_sizes; (void)n_in; (void)out_size; (void)ws_size;

    // ---- workspace carve ----
    char* base = (char*)d_ws;
    size_t off = 0;
    auto alloc = [&](size_t bytes) -> void* {
        void* p = base + off;
        off = (off + bytes + 255) & ~(size_t)255;
        return p;
    };
    const size_t ND4 = (size_t)NN * DD * 4;
    float*    h    = (float*)   alloc(ND4);
    float*    s0   = (float*)   alloc(ND4);
    float*    s1   = (float*)   alloc(ND4);
    float*    s2   = (float*)   alloc(ND4);
    float*    fsB  = (float*)   alloc(ND4);
    void*     RA   =            alloc((size_t)NN * 512 * 2);  // fd(f32)/ffh(f16)/cat(f16)
    float*    agg  = (float*)   alloc(ND4);                   // also out-head f16 scratch
    half_t*   hn   = (half_t*)  alloc((size_t)NN * DD * 2);
    float*    logi = (float*)   alloc((size_t)EE * HH * 4);
    float*    aBuf = (float*)   alloc((size_t)EE * HH * 4);
    unsigned* mEnc = (unsigned*)alloc((size_t)NN * HH * 4);
    float*    sBuf = (float*)   alloc((size_t)NN * HH * 4);
    float*    tsin = (float*)   alloc(8 * 256 * 4);
    float*    thid = (float*)   alloc(8 * 512 * 4);
    float*    te   = (float*)   alloc(8 * 256 * 4);
    float*    temb = (float*)   alloc(8 * 256 * 4);
    float*    tcAll= (float*)   alloc(NBLK * 8 * 256 * 4);
    half_t*   whS  = (half_t*)  alloc((size_t)NBLK * DD * DD * 2);
    half_t*   whD  = (half_t*)  alloc((size_t)NBLK * DD * DD * 2);
    half_t*   whF1 = (half_t*)  alloc((size_t)NBLK * DD * 2 * DD * 2);
    half_t*   whF2 = (half_t*)  alloc((size_t)NBLK * 2 * DD * DD * 2);
    half_t*   whIP = (half_t*)  alloc((size_t)(INID + DD) * DD * 2);
    half_t*   whO1 = (half_t*)  alloc((size_t)DD * DD * 2);
    half_t*   whO2 = (half_t*)  alloc((size_t)DD * INID * 2);

    float* fd  = (float*)RA;
    half_t* ffh = (half_t*)RA;
    half_t* cat = (half_t*)RA;
    half_t* oh  = (half_t*)agg;   // out-head f16 intermediate (agg dead by then)
    float* dout = (float*)d_out;

    auto cv = [&](const float* in, half_t* out, int n) {
        k_f32_to_f16<<<(n + 255) / 256, 256, 0, stream>>>(in, out, n);
    };
    // ---- weight conversion to f16 ----
    cv(wsrc, whS,  NBLK * DD * DD);
    cv(wdst, whD,  NBLK * DD * DD);
    cv(fw1,  whF1, NBLK * DD * 2 * DD);
    cv(fw2,  whF2, NBLK * 2 * DD * DD);
    cv(ip_w, whIP, (INID + DD) * DD);
    cv(o_w1, whO1, DD * DD);
    cv(o_w2, whO2, DD * INID);

    // ---- time embedding (tiny, naive f32) ----
    k_time_sin<<<8, 256, 0, stream>>>(tidx, tsin);
    k_small_gemm<0, 1><<<8 * 512 / 256, 256, 0, stream>>>(tsin, tp_w1, tp_b1, thid, 256, 512);
    k_small_gemm<0, 0><<<8 * 256 / 256, 256, 0, stream>>>(thid, tp_w2, tp_b2, te,   512, 256);
    k_small_gemm<2, 0><<<8 * 256 / 256, 256, 0, stream>>>(te,   tm_w,  tm_b,  temb, 256, 256);
    for (int b = 0; b < NBLK; ++b)
        k_small_gemm<0, 1><<<8 * 256 / 256, 256, 0, stream>>>(
            temb, tw + (size_t)b * DD * DD, tb + (size_t)b * DD,
            tcAll + (size_t)b * 8 * DD, 256, 256);

    const int NE = NN * DD;                 // N*D elementwise size
    // ---- input projection ----
    k_concat_f16<<<NN * (INID + DD) / 256, 256, 0, stream>>>(x_t, temb, cat);
    {
        dim3 g(NN / 128, DD / 64);
        k_gemm<1, true, false, false><<<g, 256, 0, stream>>>(
            cat, whIP, ip_b, h, nullptr, nullptr, NN, INID + DD, DD);
    }
    hipMemcpyAsync(s0, h, ND4, hipMemcpyDeviceToDevice, stream);

    // ---- 6 residual GAT blocks ----
    for (int b = 0; b < NBLK; ++b) {
        if (b == 3) k_add<<<NE / 256, 256, 0, stream>>>(h, s2);
        if (b == 4) k_add<<<NE / 256, 256, 0, stream>>>(h, s1);
        if (b == 5) k_add<<<NE / 256, 256, 0, stream>>>(h, s0);

        // LN1 -> f16
        k_layernorm_f16<<<NN / 8, 256, 0, stream>>>(
            h, ln1_g + b * DD, ln1_b + b * DD, hn);
        // fs, fd GEMMs (WMMA)
        dim3 gD(NN / 128, DD / 64);
        k_gemm<0, true, false, false><<<gD, 256, 0, stream>>>(
            hn, whS + (size_t)b * DD * DD, bsrc + b * DD,
            fsB, nullptr, nullptr, NN, DD, DD);
        k_gemm<0, true, false, false><<<gD, 256, 0, stream>>>(
            hn, whD + (size_t)b * DD * DD, bdst + b * DD,
            fd, nullptr, nullptr, NN, DD, DD);
        // init segment buffers: m = enc(-inf) = 0x007FFFFF, s = 0, agg = 0
        k_fill_u32<<<NN * HH / 256, 256, 0, stream>>>(mEnc, 0x007FFFFFu, NN * HH);
        k_fill_u32<<<NN * HH / 256, 256, 0, stream>>>((unsigned*)sBuf, 0u, NN * HH);
        k_fill_u32<<<NE / 256, 256, 0, stream>>>((unsigned*)agg, 0u, NE);
        // edge softmax + aggregation
        k_edge_logits<<<EE / 8, 256, 0, stream>>>(
            fsB, fd, src, dst, attn + (size_t)b * DD, logi, mEnc);
        k_edge_exp<<<EE * HH / 256, 256, 0, stream>>>(logi, mEnc, dst, aBuf, sBuf);
        k_edge_scatter<<<EE / 8, 256, 0, stream>>>(fsB, src, dst, aBuf, sBuf, agg);
        // h = gelu(agg) + tc[graph] + h
        k_gat_post<<<NE / 256, 256, 0, stream>>>(h, agg, tcAll + (size_t)b * 8 * DD);
        // FFN: LN2 -> f16, [256->512] gelu f16, [512->256] + residual into h
        k_layernorm_f16<<<NN / 8, 256, 0, stream>>>(
            h, ln2_g + b * DD, ln2_b + b * DD, hn);
        dim3 g1(NN / 128, (2 * DD) / 64);
        k_gemm<1, false, true, false><<<g1, 256, 0, stream>>>(
            hn, whF1 + (size_t)b * DD * 2 * DD, fb1 + b * 2 * DD,
            nullptr, ffh, nullptr, NN, DD, 2 * DD);
        k_gemm<0, true, false, true><<<gD, 256, 0, stream>>>(
            ffh, whF2 + (size_t)b * 2 * DD * DD, fb2 + b * DD,
            h, nullptr, h, NN, 2 * DD, DD);

        if (b == 0) hipMemcpyAsync(s1, h, ND4, hipMemcpyDeviceToDevice, stream);
        if (b == 1) hipMemcpyAsync(s2, h, ND4, hipMemcpyDeviceToDevice, stream);
    }

    // ---- outputs: (denoised [N,128], h [N,256]) concatenated ----
    hipMemcpyAsync(dout + (size_t)NN * INID, h, ND4, hipMemcpyDeviceToDevice, stream);
    k_layernorm_f16<<<NN / 8, 256, 0, stream>>>(h, oln_g, oln_b, hn);
    {
        dim3 g1(NN / 128, DD / 64);
        k_gemm<1, false, true, false><<<g1, 256, 0, stream>>>(
            hn, whO1, o_b1, nullptr, oh, nullptr, NN, DD, DD);
        dim3 g2(NN / 128, INID / 64);
        k_gemm<0, true, false, false><<<g2, 256, 0, stream>>>(
            oh, whO2, o_b2, dout, nullptr, nullptr, NN, DD, INID);
    }
}
